// EFTemporalTransformer_42356967473696
// MI455X (gfx1250) — compile-verified, matrix operand order checked by reading, not run on hardware
//
#include <hip/hip_runtime.h>

typedef unsigned short u16;
typedef unsigned int   u32;
typedef __attribute__((ext_vector_type(16))) __bf16 v16bf;
typedef __attribute__((ext_vector_type(8)))  float  v8f;
typedef __attribute__((ext_vector_type(4)))  u32    u32x4;
typedef __attribute__((ext_vector_type(2)))  u32    u32x2;
typedef __attribute__((ext_vector_type(2)))  int    i32x2;

static_assert(sizeof(v16bf) == 32, "v16bf must be 32 bytes");

// CDNA5 async global->LDS staging (ASYNCcnt-tracked), guarded so absence of the
// builtin falls back to a register bounce without breaking the compile.
#if defined(__gfx1250__) && __has_builtin(__builtin_amdgcn_global_load_async_to_lds_b64) && __has_builtin(__builtin_amdgcn_s_wait_asynccnt)
#define USE_ASYNC_LDS 1
// builtin signature (from hipcc diagnostic): (V2i AS1*, V2i AS3*, imm, imm)
typedef __attribute__((address_space(1))) i32x2 gi32x2;
typedef __attribute__((address_space(3))) i32x2 li32x2;
#define ASYNC_WAIT(n) __builtin_amdgcn_s_wait_asynccnt(n)
#else
#define USE_ASYNC_LDS 0
#define ASYNC_WAIT(n) ((void)0)
#endif

// ---- problem sizes (compile-time) ----
constexpr int B_  = 16384;
constexpr int S_  = 4;
constexpr int T_  = B_ * S_;   // 65536 token rows
constexpr int E_  = 1152;
constexpr int D_  = 192;
constexpr int H_  = 4;
constexpr int DH_ = 48;
constexpr int FF_ = 768;
constexpr int NL_ = 2;

__device__ __forceinline__ u16 f2bf(float f) {
  u32 u = __float_as_uint(f);
  u += 0x7FFFu + ((u >> 16) & 1u);          // round-to-nearest-even
  return (u16)(u >> 16);
}
__device__ __forceinline__ float bf2f(u16 h) {
  return __uint_as_float(((u32)h) << 16);
}
__device__ __forceinline__ float gelu_exact(float x) {
  return 0.5f * x * (1.0f + erff(x * 0.70710678118654752f));
}

enum { EPI_BF16 = 0, EPI_GELU_BF16 = 1, EPI_GELU_PE_F32 = 2, EPI_RESID_F32 = 3 };

union ABOp { u32x4 u[2]; v16bf v; };

// ------------------------------------------------------------------
// Tiled bf16 WMMA GEMM:  out[M,N] (+)= epi( A[M,K] @ W[N,K]^T + bias )
// Block = 8 waves; block owns one 16-row M tile and up to 8 N tiles.
// A staged global->LDS with CDNA5 async copies, 3-deep pipeline,
// one barrier per 64-K chunk (2 WMMAs). W streamed from L2 per wave.
// Pipeline drain is peeled so every wait immediate is compile-time
// and the steady-state body is branch-free.
// ------------------------------------------------------------------
template <int EPI>
__global__ __launch_bounds__(256)
void wmma_gemm(const u16* __restrict__ Amat, const u16* __restrict__ Wmat,
               const float* __restrict__ bias, void* __restrict__ outp,
               const float* __restrict__ pe, int N, int K) {
  const int mtile  = blockIdx.x;
  const int wave   = threadIdx.x >> 5;     // wave32
  const int lane   = threadIdx.x & 31;
  const int ntiles = N >> 4;

  int ntile = blockIdx.y * 8 + wave;
  const bool active = (ntile < ntiles);
  if (!active) ntile = ntiles - 1;         // clamp: keep barriers uniform

  // 3 ping-pong buffers, each one 16x64 bf16 A chunk (2 KB)
  __shared__ u16 As[3][16 * 64];

  const int row  = lane & 15;              // tile row (A) / tile col (B)
  const int koff = (lane >> 4) << 3;       // 0 or 8 per ISA 16-bit layout

  const u16* Abase = Amat + (size_t)mtile * 16 * K;
  const u16* Wrow  = Wmat + ((size_t)ntile * 16 + row) * K;

  // per-thread staging slice: 4 bf16 (8 bytes) of the 16x64 chunk
  const int  sr   = threadIdx.x >> 4;            // 0..15  (tile row)
  const int  sc   = (threadIdx.x & 15) << 2;     // 0..60  (col, x4 elems)
  const u16* Asrc = Abase + (size_t)sr * K + sc;
  u16*       Ldst = &As[0][0] + (sr * 64 + sc);  // +1024 per buffer

  const int nch = K >> 6;                  // 64-K chunks (all K are multiples)

  auto stage = [&](int c, int buf) {
#if USE_ASYNC_LDS
    __builtin_amdgcn_global_load_async_to_lds_b64(
        (gi32x2*)(Asrc + (c << 6)),
        (li32x2*)(Ldst + buf * 1024),
        0, 0);
#else
    *(u32x2*)(Ldst + buf * 1024) = *(const u32x2*)(Asrc + (c << 6));
#endif
  };

  v8f acc = {};
  auto compute_chunk = [&](int c, int buf) {
    const int k0 = c << 6;
    __builtin_prefetch(Wrow + k0 + 128, 0, 3);   // weight stream prefetch
    const u16* cb = &As[buf][0];
    ABOp a0, a1, b0, b1;
    a0.u[0] = *(const u32x4*)(cb + row * 64 + koff);
    a0.u[1] = *(const u32x4*)(cb + row * 64 + koff + 16);
    a1.u[0] = *(const u32x4*)(cb + row * 64 + 32 + koff);
    a1.u[1] = *(const u32x4*)(cb + row * 64 + 32 + koff + 16);
    b0.u[0] = *(const u32x4*)(Wrow + k0 + koff);
    b0.u[1] = *(const u32x4*)(Wrow + k0 + koff + 16);
    b1.u[0] = *(const u32x4*)(Wrow + k0 + 32 + koff);
    b1.u[1] = *(const u32x4*)(Wrow + k0 + 32 + koff + 16);
    acc = __builtin_amdgcn_wmma_f32_16x16x32_bf16(false, a0.v, false, b0.v,
                                                  (short)0, acc, false, false);
    acc = __builtin_amdgcn_wmma_f32_16x16x32_bf16(false, a1.v, false, b1.v,
                                                  (short)0, acc, false, false);
  };

  // prologue: chunks 0 and 1 in flight
  stage(0, 0);
  if (nch > 1) stage(1, 1);

  // steady state: wait<=1 (in-order async => oldest chunk done), stage c+2
  int cur = 0;
  for (int c = 0; c + 2 < nch; ++c) {
    ASYNC_WAIT(1);
    __syncthreads();                       // publish chunk c; drains DS reads
    int nxt = cur + 2; if (nxt >= 3) nxt -= 3;
    stage(c + 2, nxt);                     // overwrite-safe after barrier
    compute_chunk(c, cur);
    cur += 1; if (cur >= 3) cur -= 3;
  }
  // chunk nch-2: nothing left to stage
  if (nch > 1) {
    ASYNC_WAIT(1);
    __syncthreads();
    compute_chunk(nch - 2, cur);
    cur += 1; if (cur >= 3) cur -= 3;
  }
  // final chunk: drain
  ASYNC_WAIT(0);
  __syncthreads();
  compute_chunk(nch - 1, cur);

  if (!active) return;
  // C/D layout: VGPR r, lanes 0-15 -> M=r, lanes 16-31 -> M=r+8; N = lane%16
  const int n     = (ntile << 4) + (lane & 15);
  const int mbase = (mtile << 4) + ((lane >> 4) << 3);
  const float bv  = bias[n];
#pragma unroll
  for (int r = 0; r < 8; ++r) {
    const int m = mbase + r;
    float v = acc[r] + bv;
    const size_t idx = (size_t)m * N + n;
    if (EPI == EPI_BF16) {
      ((u16*)outp)[idx] = f2bf(v);
    } else if (EPI == EPI_GELU_BF16) {
      ((u16*)outp)[idx] = f2bf(gelu_exact(v));
    } else if (EPI == EPI_GELU_PE_F32) {
      ((float*)outp)[idx] = gelu_exact(v) + pe[(m & 3) * N + n];
    } else { // EPI_RESID_F32
      ((float*)outp)[idx] += v;
    }
  }
}

// ---- LayerNorm: one wave per row, fp32 in -> bf16 out ----
__global__ __launch_bounds__(256)
void ln_rows(const float* __restrict__ x, const float* __restrict__ g,
             const float* __restrict__ be, u16* __restrict__ out,
             int rows, int width) {
  const int wave = threadIdx.x >> 5, lane = threadIdx.x & 31;
  const int row = blockIdx.x * 8 + wave;
  if (row >= rows) return;
  const float* xr = x + (size_t)row * width;
  float s = 0.f, ss = 0.f;
  for (int i = lane; i < width; i += 32) { float v = xr[i]; s += v; ss += v * v; }
#pragma unroll
  for (int off = 16; off >= 1; off >>= 1) {
    s  += __shfl_xor(s,  off, 32);
    ss += __shfl_xor(ss, off, 32);
  }
  const float mean = s / (float)width;
  const float var  = ss / (float)width - mean * mean;
  const float rstd = rsqrtf(var + 1e-5f);
  u16* orow = out + (size_t)row * width;
  for (int i = lane; i < width; i += 32)
    orow[i] = f2bf((xr[i] - mean) * rstd * g[i] + be[i]);
}

// ---- tiny S=4 attention: one thread per (b, h, s_q) ----
__global__ __launch_bounds__(256)
void attn_kernel(const u16* __restrict__ qkv, u16* __restrict__ o) {
  const int t = blockIdx.x * 256 + threadIdx.x;    // over B*H*S
  const int sq = t & 3, h = (t >> 2) & 3, b = t >> 4;
  const u16* base = qkv + (size_t)b * 4 * (3 * D_);
  const u16* qrow = base + (size_t)sq * (3 * D_) + h * DH_;
  float sc[4]; float mx = -3.0e38f;
#pragma unroll
  for (int sk = 0; sk < 4; ++sk) {
    const u16* krow = base + (size_t)sk * (3 * D_) + D_ + h * DH_;
    float d = 0.f;
    for (int j = 0; j < DH_; ++j) d += bf2f(qrow[j]) * bf2f(krow[j]);
    d *= 0.14433756729740643f;                     // 1/sqrt(48)
    sc[sk] = d; mx = fmaxf(mx, d);
  }
  float den = 0.f;
#pragma unroll
  for (int sk = 0; sk < 4; ++sk) { sc[sk] = __expf(sc[sk] - mx); den += sc[sk]; }
  const float inv = 1.f / den;
  u16* orow = o + ((size_t)(b * 4 + sq)) * D_ + h * DH_;
  for (int j = 0; j < DH_; ++j) {
    float a = 0.f;
#pragma unroll
    for (int sk = 0; sk < 4; ++sk)
      a += sc[sk] * bf2f(base[(size_t)sk * (3 * D_) + 2 * D_ + h * DH_ + j]);
    orow[j] = f2bf(a * inv);
  }
}

// ---- mean pool over S=4 ----
__global__ __launch_bounds__(256)
void pool_kernel(const float* __restrict__ x, float* __restrict__ pooled) {
  const int t = blockIdx.x * 256 + threadIdx.x;    // over B*D
  const int b = t / D_, d = t - b * D_;
  const float* r = x + (size_t)b * 4 * D_ + d;
  pooled[t] = 0.25f * (r[0] + r[D_] + r[2 * D_] + r[3 * D_]);
}

// ---- final 64->1 head ----
__global__ __launch_bounds__(256)
void head3_kernel(const u16* __restrict__ h2o, const float* __restrict__ w,
                  const float* __restrict__ b, float* __restrict__ out) {
  const int t = blockIdx.x * 256 + threadIdx.x;    // over B
  const u16* r = h2o + (size_t)t * 64;
  float acc = 0.f;
#pragma unroll
  for (int j = 0; j < 64; ++j) acc += bf2f(r[j]) * w[j];
  out[t] = acc + b[0];
}

// ---- sinusoidal positional encoding (S_ x D_) ----
__global__ void pe_kernel(float* __restrict__ pe) {
  const int t = threadIdx.x;                       // 0..767
  if (t >= S_ * D_) return;
  const int s = t / D_, d = t - s * D_;
  const float div = __expf(-((float)(d & ~1)) * (__logf(10000.f) / (float)D_));
  const float ang = (float)s * div;
  pe[t] = (d & 1) ? __cosf(ang) : __sinf(ang);
}

// ---- fp32 -> bf16 weight conversion ----
__global__ __launch_bounds__(256)
void cvt_kernel(const float* __restrict__ in, u16* __restrict__ out, int n) {
  const int t = blockIdx.x * 256 + threadIdx.x;
  if (t < n) out[t] = f2bf(in[t]);
}

static inline size_t al256(size_t b) { return (b + 255) & ~(size_t)255; }

extern "C" void kernel_launch(void* const* d_in, const int* in_sizes, int n_in,
                              void* d_out, int out_size, void* d_ws, size_t ws_size,
                              hipStream_t stream) {
  (void)in_sizes; (void)n_in; (void)out_size; (void)ws_size;
  const float* emb       = (const float*)d_in[0];
  const float* in_ln_w   = (const float*)d_in[1];
  const float* in_ln_b   = (const float*)d_in[2];
  const float* in_proj_w = (const float*)d_in[3];
  const float* in_proj_b = (const float*)d_in[4];
  const float* qkv_w     = (const float*)d_in[5];
  const float* qkv_b     = (const float*)d_in[6];
  const float* ao_w      = (const float*)d_in[7];
  const float* ao_b      = (const float*)d_in[8];
  const float* ln1_w     = (const float*)d_in[9];
  const float* ln1_b     = (const float*)d_in[10];
  const float* ln2_w     = (const float*)d_in[11];
  const float* ln2_b     = (const float*)d_in[12];
  const float* ff1_w     = (const float*)d_in[13];
  const float* ff1_b     = (const float*)d_in[14];
  const float* ff2_w     = (const float*)d_in[15];
  const float* ff2_b     = (const float*)d_in[16];
  const float* hln_w     = (const float*)d_in[17];
  const float* hln_b     = (const float*)d_in[18];
  const float* h1_w      = (const float*)d_in[19];
  const float* h1_b      = (const float*)d_in[20];
  const float* h2_w      = (const float*)d_in[21];
  const float* h2_b      = (const float*)d_in[22];
  const float* h3_w      = (const float*)d_in[23];
  const float* h3_b      = (const float*)d_in[24];
  float* out = (float*)d_out;

  // ---- workspace layout ----
  char* ws = (char*)d_ws;
  size_t off = 0;
  auto alloc = [&](size_t bytes) -> char* {
    char* p = ws + off; off += al256(bytes); return p;
  };
  u16*  wb_inproj = (u16*)alloc((size_t)D_ * E_ * 2);
  u16*  wb_qkv    = (u16*)alloc((size_t)NL_ * 3 * D_ * D_ * 2);
  u16*  wb_ao     = (u16*)alloc((size_t)NL_ * D_ * D_ * 2);
  u16*  wb_ff1    = (u16*)alloc((size_t)NL_ * FF_ * D_ * 2);
  u16*  wb_ff2    = (u16*)alloc((size_t)NL_ * D_ * FF_ * 2);
  u16*  wb_h1     = (u16*)alloc((size_t)256 * D_ * 2);
  u16*  wb_h2     = (u16*)alloc((size_t)64 * 256 * 2);
  float* pe       = (float*)alloc((size_t)S_ * D_ * 4);
  float* x        = (float*)alloc((size_t)T_ * D_ * 4);   // persistent fp32 residual stream
  char*  big      = ws + off;                              // reused region
  // phase A: normalized embeddings
  u16*  xn_e   = (u16*)big;                                           // T x E
  // phase B (per layer): xn | qkv | o   (ffh overlays qkv after attn done)
  u16*  xn     = (u16*)big;                                           // T x D
  u16*  qkvb   = (u16*)(big + al256((size_t)T_ * D_ * 2));            // T x 3D
  u16*  obuf   = (u16*)(big + al256((size_t)T_ * D_ * 2)
                            + al256((size_t)T_ * 3 * D_ * 2));        // T x D
  u16*  ffh    = qkvb;                                                // T x FF
  // phase C (head)
  float* pooled  = (float*)big;                                       // B x D
  u16*   pooledn = (u16*)(big + al256((size_t)B_ * D_ * 4));          // B x D
  u16*   h1o     = (u16*)((char*)pooledn + al256((size_t)B_ * D_ * 2));
  u16*   h2o     = (u16*)((char*)h1o + al256((size_t)B_ * 256 * 2));

  // ---- weight conversion + posenc ----
  auto cvt = [&](const float* src, u16* dst, size_t n) {
    cvt_kernel<<<(unsigned)((n + 255) / 256), 256, 0, stream>>>(src, dst, (int)n);
  };
  cvt(in_proj_w, wb_inproj, (size_t)D_ * E_);
  cvt(qkv_w,     wb_qkv,    (size_t)NL_ * 3 * D_ * D_);
  cvt(ao_w,      wb_ao,     (size_t)NL_ * D_ * D_);
  cvt(ff1_w,     wb_ff1,    (size_t)NL_ * FF_ * D_);
  cvt(ff2_w,     wb_ff2,    (size_t)NL_ * D_ * FF_);
  cvt(h1_w,      wb_h1,     (size_t)256 * D_);
  cvt(h2_w,      wb_h2,     (size_t)64 * 256);
  pe_kernel<<<1, S_ * D_, 0, stream>>>(pe);

  // ---- input: LN(E) -> bf16, then in_proj GEMM (+bias, GELU, +posenc) -> x fp32 ----
  ln_rows<<<T_ / 8, 256, 0, stream>>>(emb, in_ln_w, in_ln_b, xn_e, T_, E_);
  wmma_gemm<EPI_GELU_PE_F32><<<dim3(T_ / 16, (D_ / 16 + 7) / 8), 256, 0, stream>>>(
      xn_e, wb_inproj, in_proj_b, x, pe, D_, E_);

  // ---- 2 pre-norm transformer layers ----
  for (int i = 0; i < NL_; ++i) {
    ln_rows<<<T_ / 8, 256, 0, stream>>>(x, ln1_w + i * D_, ln1_b + i * D_, xn, T_, D_);
    wmma_gemm<EPI_BF16><<<dim3(T_ / 16, (3 * D_ / 16 + 7) / 8), 256, 0, stream>>>(
        xn, wb_qkv + (size_t)i * 3 * D_ * D_, qkv_b + i * 3 * D_, qkvb, nullptr,
        3 * D_, D_);
    attn_kernel<<<(B_ * H_ * S_) / 256, 256, 0, stream>>>(qkvb, obuf);
    wmma_gemm<EPI_RESID_F32><<<dim3(T_ / 16, (D_ / 16 + 7) / 8), 256, 0, stream>>>(
        obuf, wb_ao + (size_t)i * D_ * D_, ao_b + i * D_, x, nullptr, D_, D_);
    ln_rows<<<T_ / 8, 256, 0, stream>>>(x, ln2_w + i * D_, ln2_b + i * D_, xn, T_, D_);
    wmma_gemm<EPI_GELU_BF16><<<dim3(T_ / 16, (FF_ / 16 + 7) / 8), 256, 0, stream>>>(
        xn, wb_ff1 + (size_t)i * FF_ * D_, ff1_b + i * FF_, ffh, nullptr, FF_, D_);
    wmma_gemm<EPI_RESID_F32><<<dim3(T_ / 16, (D_ / 16 + 7) / 8), 256, 0, stream>>>(
        ffh, wb_ff2 + (size_t)i * D_ * FF_, ff2_b + i * D_, x, nullptr, D_, FF_);
  }

  // ---- head: mean pool -> LN -> 192x256 GELU -> 256x64 GELU -> 64x1 ----
  pool_kernel<<<(B_ * D_) / 256, 256, 0, stream>>>(x, pooled);
  ln_rows<<<B_ / 8, 256, 0, stream>>>(pooled, hln_w, hln_b, pooledn, B_, D_);
  wmma_gemm<EPI_GELU_BF16><<<dim3(B_ / 16, 2), 256, 0, stream>>>(
      pooledn, wb_h1, h1_b, h1o, nullptr, 256, D_);
  wmma_gemm<EPI_GELU_BF16><<<dim3(B_ / 16, 1), 256, 0, stream>>>(
      h1o, wb_h2, h2_b, h2o, nullptr, 64, 256);
  head3_kernel<<<B_ / 256, 256, 0, stream>>>(h2o, h3_w, h3_b, out);
}
